// fuckupnet_15066745274659
// MI455X (gfx1250) — compile-verified, matrix operand order checked by reading, not run on hardware
//
#include <hip/hip_runtime.h>
#include <hip/hip_bf16.h>
#include <math.h>

// ---------------------------------------------------------------------------
// Types for CDNA5 WMMA (wave32): D[16x16,f32] = A[16x32,bf16] x B[32x16,bf16] + C
// ---------------------------------------------------------------------------
typedef __attribute__((ext_vector_type(16))) __bf16 v16bf;
typedef __attribute__((ext_vector_type(8)))  float  v8f;

#define FEAT_PAD 160   // 144 real channels, padded to 5 * 32 for K-steps
#define NPRE 1024
#define NCUR 1024

// |x| via an optimizer-opaque VALU sign-bit clear. Plain fabsf / maxnum(x,-x)
// get canonicalized to llvm.fabs, which SLP vectorizes to fabs.v2f32 and the
// gfx1250 backend crashes on (unselectable v2f32 bitcast/and). Inline asm is
// immune to that fold and blocks SLP across this chain.
__device__ __forceinline__ float abs_opaque(float x) {
    float r;
    asm("v_and_b32 %0, 0x7fffffff, %1" : "=v"(r) : "v"(x));
    return r;
}

// ===========================================================================
// Kernel A: per-image feature extraction (conv1+conv2 fused via rolling rows,
// GAP, app fc, motion MLP). One 256-thread block per image; 2048 blocks.
// Writes features [2048][160] f32 into workspace (pad channels zeroed).
// ===========================================================================
__global__ __launch_bounds__(256)
void feat_kernel(const float* __restrict__ pre_crop, const float* __restrict__ cur_crop,
                 const float* __restrict__ pre_motion, const float* __restrict__ cur_motion,
                 const float* __restrict__ conv1_w, const float* __restrict__ conv1_b,
                 const float* __restrict__ conv2_w, const float* __restrict__ conv2_b,
                 const float* __restrict__ app_fc_w, const float* __restrict__ app_fc_b,
                 const float* __restrict__ mot_fc1_w, const float* __restrict__ mot_fc1_b,
                 const float* __restrict__ mot_fc2_w, const float* __restrict__ mot_fc2_b,
                 float* __restrict__ feat_out)
{
    __shared__ float s_in[3 * 84 * 32];     // staged input image (31.5 KB)
    __shared__ float s_c1[3][32][16];       // rolling 3 conv1 rows (6 KB)
    __shared__ float s_cw[32 * 3 * 3 * 3];  // conv1 weights
    __shared__ float s_cb[32];
    __shared__ float s_pool[64];
    __shared__ float s_m1[32];

    const int t   = threadIdx.x;
    const int img = blockIdx.x;                 // 0..2047 (pre then cur)
    const bool isPre = (img < NPRE);
    const int  n     = isPre ? img : img - NPRE;
    const float* crop   = (isPre ? pre_crop   : cur_crop)   + (size_t)n * (3 * 84 * 32);
    const float* motion = (isPre ? pre_motion : cur_motion) + (size_t)n * 4;
    float* fout = feat_out + (size_t)img * FEAT_PAD;

    for (int i = t; i < 3 * 84 * 32; i += 256) s_in[i] = crop[i];
    for (int i = t; i < 864;         i += 256) s_cw[i] = conv1_w[i];
    if (t < 32) s_cb[t] = conv1_b[t];
    __syncthreads();

    // conv2 output (oc,ow) ownership: idx=t -> (t>>3, t&7); idx=t+256 -> (+32, same ow)
    const int ocA = t >> 3;
    const int owA = t & 7;
    float accA = 0.f, accB = 0.f;   // GAP accumulators for oc and oc+32

    int nextRow = 0;                // next conv1 row to produce (0..41)
    for (int o2 = 0; o2 < 21; ++o2) {
        const int need = 2 * o2 + 1;
        while (nextRow <= need) {
            __syncthreads();        // previous conv2 row done before slot reuse
            for (int idx = t; idx < 512; idx += 256) {
                const int c = idx >> 4, ow = idx & 15;
                float a = s_cb[c];
                const int r0 = 2 * nextRow - 1;
                const int cbase = 2 * ow - 1;
#pragma unroll
                for (int kh = 0; kh < 3; ++kh) {
                    const int r = r0 + kh;
                    if (r < 0 || r >= 84) continue;
#pragma unroll
                    for (int kw = 0; kw < 3; ++kw) {
                        const int cc = cbase + kw;
                        if (cc < 0 || cc >= 32) continue;
#pragma unroll
                        for (int ic = 0; ic < 3; ++ic)
                            a += s_in[(ic * 84 + r) * 32 + cc] *
                                 s_cw[((c * 3 + ic) * 3 + kh) * 3 + kw];
                    }
                }
                s_c1[nextRow % 3][c][ow] = fmaxf(a, 0.f);
            }
            ++nextRow;
        }
        __syncthreads();
        // conv2 row o2
        float a0 = conv2_b[ocA];
        float a1 = conv2_b[ocA + 32];
        const int r0 = 2 * o2 - 1;
        const int cbase = 2 * owA - 1;
        for (int ic = 0; ic < 32; ++ic) {
#pragma unroll
            for (int kh = 0; kh < 3; ++kh) {
                const int r = r0 + kh;
                if (r < 0 || r >= 42) continue;
                const float* crow = &s_c1[r % 3][ic][0];
#pragma unroll
                for (int kw = 0; kw < 3; ++kw) {
                    const int cc = cbase + kw;
                    if (cc < 0 || cc >= 16) continue;
                    const float v = crow[cc];
                    a0 += v * conv2_w[(( ocA       * 32 + ic) * 3 + kh) * 3 + kw];
                    a1 += v * conv2_w[(((ocA + 32) * 32 + ic) * 3 + kh) * 3 + kw];
                }
            }
        }
        accA += fmaxf(a0, 0.f);
        accB += fmaxf(a1, 0.f);
    }

    // GAP: reduce over the 8 lanes sharing an oc (consecutive lanes)
#pragma unroll
    for (int off = 1; off < 8; off <<= 1) {
        accA += __shfl_xor(accA, off, 32);
        accB += __shfl_xor(accB, off, 32);
    }
    if ((t & 7) == 0) {
        s_pool[ocA]      = accA * (1.f / 168.f);
        s_pool[ocA + 32] = accB * (1.f / 168.f);
    }
    __syncthreads();

    if (t < 128) {                       // appearance fc 64 -> 128 + relu
        float a = app_fc_b[t];
#pragma unroll 8
        for (int c = 0; c < 64; ++c) a += s_pool[c] * app_fc_w[t * 64 + c];
        fout[t] = fmaxf(a, 0.f);
    }
    if (t < 32) {                        // motion fc1 4 -> 32 + relu
        float a = mot_fc1_b[t];
#pragma unroll
        for (int c = 0; c < 4; ++c) a += motion[c] * mot_fc1_w[t * 4 + c];
        s_m1[t] = fmaxf(a, 0.f);
    }
    __syncthreads();
    if (t >= 128 && t < 144) {           // motion fc2 32 -> 16 (no relu)
        const int o = t - 128;
        float a = mot_fc2_b[o];
#pragma unroll
        for (int h = 0; h < 32; ++h) a += s_m1[h] * mot_fc2_w[o * 32 + h];
        fout[t] = a;
    }
    if (t >= 144 && t < FEAT_PAD) fout[t] = 0.f;   // zero pad -> |0-0| = 0 in K-pad
}

// ===========================================================================
// Kernel B: pairwise classifier on WMMA (bf16).
// Block = 16 pre x 16 cur = 256 pairs. GEMM [16x160] x [160x64] per pre-row
// tile via v_wmma_f32_16x16x32_bf16 (5 K-steps x 4 N-tiles), 8 waves x 2
// M-tiles each. Epilogue: bias+relu, 64-h reduction to 2 logits via 16-lane
// shfl_xor butterflies, softmax, write score0 + matrix.
// ===========================================================================
__global__ __launch_bounds__(256)
void pair_kernel(const float* __restrict__ feat,   // [2048][160]: pre rows then cur rows
                 const float* __restrict__ w1,     // [64][144]
                 const float* __restrict__ b1,     // [64]
                 const float* __restrict__ w2,     // [2][64]
                 const float* __restrict__ b2,     // [2]
                 float* __restrict__ out)          // [N*M*2] score0 ++ [N*M] matrix
{
    __shared__ float  sfp[16 * FEAT_PAD];
    __shared__ float  sfc[16 * FEAT_PAD];
    __shared__ __bf16 sW1[64 * FEAT_PAD];
    __shared__ float  sB1[64];
    __shared__ float  sW2[128];
    __shared__ float  sB2[2];

    const int t  = threadIdx.x;
    const int j0 = blockIdx.x * 16;   // cur tile base
    const int i0 = blockIdx.y * 16;   // pre tile base
    const float* fpre = feat;
    const float* fcur = feat + (size_t)NPRE * FEAT_PAD;

    for (int i = t; i < 16 * FEAT_PAD; i += 256) {
        sfp[i] = fpre[(size_t)i0 * FEAT_PAD + i];
        sfc[i] = fcur[(size_t)j0 * FEAT_PAD + i];
    }
    for (int i = t; i < 64 * FEAT_PAD; i += 256) {
        const int h = i / FEAT_PAD, c = i % FEAT_PAD;
        sW1[i] = (__bf16)((c < 144) ? w1[h * 144 + c] : 0.f);
    }
    if (t < 64)  sB1[t] = b1[t];
    if (t < 128) sW2[t] = w2[t];
    if (t < 2)   sB2[t] = b2[t];
    __syncthreads();

    const int lane = t & 31;
    const int wave = t >> 5;
    const int half = lane >> 4;   // which 16-lane half
    const int l16  = lane & 15;

    // 16-bit A/B fragment K mapping (ISA 7.12.2): lanes 0-15: v0..3->K0..7,
    // v4..7->K16..23 ; lanes 16-31: +8. Element q = 2*v + e (e = low/high half).
    int kmap[16];
#pragma unroll
    for (int v = 0; v < 8; ++v)
#pragma unroll
        for (int e = 0; e < 2; ++e)
            kmap[2 * v + e] = ((v < 4) ? (2 * v + e) : (16 + 2 * (v - 4) + e)) + 8 * half;

    // Per-lane epilogue constants: h = nt*16 + l16
    float b1v[4], w2a[4], w2b[4];
#pragma unroll
    for (int nt = 0; nt < 4; ++nt) {
        b1v[nt] = sB1[nt * 16 + l16];
        w2a[nt] = sW2[nt * 16 + l16];
        w2b[nt] = sW2[64 + nt * 16 + l16];
    }

#pragma unroll
    for (int mi = 0; mi < 2; ++mi) {
        const int mt = wave * 2 + mi;                 // pre row within tile
        const float* prow = &sfp[mt * FEAT_PAD];      // fixed pre feature row
        const float* crow = &sfc[l16 * FEAT_PAD];     // A-row j = l16

        v8f acc[4] = {v8f{}, v8f{}, v8f{}, v8f{}};
#pragma unroll
        for (int ks = 0; ks < 5; ++ks) {
            const int c0 = ks * 32;
            v16bf A;
#pragma unroll
            for (int q = 0; q < 16; ++q) {
                const int c = c0 + kmap[q];
                A[q] = (__bf16)abs_opaque(prow[c] - crow[c]);  // |f_pre - f_cur| -> bf16
            }
#pragma unroll
            for (int nt = 0; nt < 4; ++nt) {
                const __bf16* wrow = &sW1[(nt * 16 + l16) * FEAT_PAD]; // B col n = l16
                v16bf B;
#pragma unroll
                for (int q = 0; q < 16; ++q) B[q] = wrow[c0 + kmap[q]];
                acc[nt] = __builtin_amdgcn_wmma_f32_16x16x32_bf16(
                    false, A, false, B, (short)0, acc[nt], false, false);
            }
        }

        // Epilogue: D layout -> VGPR r holds rows M=r (lanes 0-15) / M=r+8 (16-31),
        // column n = nt*16 + l16. Reduce 64-wide h to 2 logits per pair.
#pragma unroll
        for (int r = 0; r < 8; ++r) {
            float l0 = 0.f, l1 = 0.f;
#pragma unroll
            for (int nt = 0; nt < 4; ++nt) {
                const float hv = fmaxf(acc[nt][r] + b1v[nt], 0.f);
                l0 += w2a[nt] * hv;
                l1 += w2b[nt] * hv;
            }
#pragma unroll
            for (int off = 1; off < 16; off <<= 1) {   // stays within 16-lane halves
                l0 += __shfl_xor(l0, off, 32);
                l1 += __shfl_xor(l1, off, 32);
            }
            l0 += sB2[0];
            l1 += sB2[1];
            const float p1 = 1.f / (1.f + __expf(l0 - l1));  // softmax, 2-way
            const float p0 = 1.f - p1;
            if (l16 == 0) {
                const int jrow = r + 8 * half;
                const size_t idx = (size_t)(i0 + mt) * NCUR + (size_t)(j0 + jrow);
                out[2 * idx]     = p0;                              // score0
                out[2 * idx + 1] = p1;
                out[(size_t)2 * NPRE * NCUR + idx] = p1;            // matrix
            }
        }
    }
}

// ===========================================================================
extern "C" void kernel_launch(void* const* d_in, const int* in_sizes, int n_in,
                              void* d_out, int out_size, void* d_ws, size_t ws_size,
                              hipStream_t stream) {
    const float* pre_crop   = (const float*)d_in[0];
    const float* cur_crop   = (const float*)d_in[1];
    const float* pre_motion = (const float*)d_in[2];
    const float* cur_motion = (const float*)d_in[3];
    const float* conv1_w    = (const float*)d_in[4];
    const float* conv1_b    = (const float*)d_in[5];
    const float* conv2_w    = (const float*)d_in[6];
    const float* conv2_b    = (const float*)d_in[7];
    const float* app_fc_w   = (const float*)d_in[8];
    const float* app_fc_b   = (const float*)d_in[9];
    const float* mot_fc1_w  = (const float*)d_in[10];
    const float* mot_fc1_b  = (const float*)d_in[11];
    const float* mot_fc2_w  = (const float*)d_in[12];
    const float* mot_fc2_b  = (const float*)d_in[13];
    const float* cls_fc1_w  = (const float*)d_in[14];
    const float* cls_fc1_b  = (const float*)d_in[15];
    const float* cls_fc2_w  = (const float*)d_in[16];
    const float* cls_fc2_b  = (const float*)d_in[17];

    float* feat_ws = (float*)d_ws;   // [2048][160] f32 = 1.31 MB

    feat_kernel<<<2048, 256, 0, stream>>>(
        pre_crop, cur_crop, pre_motion, cur_motion,
        conv1_w, conv1_b, conv2_w, conv2_b, app_fc_w, app_fc_b,
        mot_fc1_w, mot_fc1_b, mot_fc2_w, mot_fc2_b, feat_ws);

    dim3 grid(NCUR / 16, NPRE / 16);   // 64 x 64 blocks
    pair_kernel<<<grid, 256, 0, stream>>>(
        feat_ws, cls_fc1_w, cls_fc1_b, cls_fc2_w, cls_fc2_b, (float*)d_out);
}
